// SplineNet_17231408791938
// MI455X (gfx1250) — compile-verified, matrix operand order checked by reading, not run on hardware
//
#include <hip/hip_runtime.h>
#include <hip/hip_bf16.h>
#include <math.h>

// SplineConv GNN, 2 layers, on gfx1250 (MI455X).
// Node-first GEMMs via v_wmma_f32_16x16x32_bf16 (fp32 accum) with weights
// pre-swizzled into B-fragment order; edge gather/blend/scatter with one
// wave32 per edge + f32 global atomics.

typedef __attribute__((ext_vector_type(16))) __bf16 v16bf;
typedef __attribute__((ext_vector_type(8)))  float  v8f;

#define NNODES 100000   // multiple of 16 -> no row guards in GEMM stores
#define NEDGES 1600000
#define FIN    128
#define HID    128
#define NCLS   40

// ---------------------------------------------------------------------------
// A fragment: 16x32 bf16 tile of row-major fp32 X, rows row0..row0+15,
// K = k0..k0+31. Documented 16-bit A layout:
//   lanes 0-15  : M=lane,    VGPR0..3 = K 0..7 (pk2),  VGPR4..7 = K 16..23
//   lanes 16-31 : M=lane-16, VGPR0..3 = K 8..15,       VGPR4..7 = K 24..31
// Converted fp32->bf16 in-register (X is read exactly once, so a separate
// conversion pass would only add bandwidth).
// ---------------------------------------------------------------------------
__device__ __forceinline__ v16bf load_a_frag(const float* __restrict__ X,
                                             int row, int ldx, int k0, int lane) {
    const int half = lane >> 4;
    const float* p = X + (size_t)row * ldx + (k0 + half * 8);
    v16bf a;
#pragma unroll
    for (int i = 0; i < 8; ++i) a[i] = (__bf16)p[i];
#pragma unroll
    for (int i = 0; i < 8; ++i) a[8 + i] = (__bf16)p[16 + i];
    return a;
}

// ---------------------------------------------------------------------------
// One-time weight pre-swizzle: fp32 (K=128 x ncols) row-major -> bf16
// B fragments laid out [ntile][kchunk][lane][16], 32 B per lane, so the GEMM
// hot loop loads each fragment as one aligned v16bf (2x b128, no cvt VALU).
// B-fragment layout mirrors A: lane = column within tile, half-waves split K.
// ---------------------------------------------------------------------------
__global__ void pack_w(const float* __restrict__ W, __bf16* __restrict__ out,
                       int ncols, int ntiles) {
    const int t = blockIdx.x * blockDim.x + threadIdx.x;
    const int nthreads = ntiles * 4 * 32;          // (ntile,kchunk) frags x 32 lanes
    if (t >= nthreads) return;
    const int frag = t >> 5;
    const int lane = t & 31;
    const int nt = frag >> 2;                      // ntile
    const int kc = frag & 3;                       // kchunk (K step 32)
    const int n  = nt * 16 + (lane & 15);
    const int kb = kc * 32 + (lane >> 4) * 8;
    __bf16* o = out + (size_t)t * 16;
#pragma unroll
    for (int i = 0; i < 8; ++i) {
        o[i]     = (__bf16)((n < ncols) ? W[(size_t)(kb + i)      * ncols + n] : 0.0f);
        o[8 + i] = (__bf16)((n < ncols) ? W[(size_t)(kb + 16 + i) * ncols + n] : 0.0f);
    }
}

// ---------------------------------------------------------------------------
// Triple GEMM: Y0 = X@W0, Y1 = X@W1, Yr = X@Wr + bias.  K = 128 fixed.
// One wave per 16-row strip; A fragments (4 K-chunks) loaded once and reused
// for all 3*NTILES output tiles -> X read exactly once from HBM. B fragments
// come pre-swizzled in bf16 (L2-resident, vector loads only). nrows must be
// a multiple of 16 (true here), so stores are unconditional with immediate
// offsets off one base pointer per tile.
// ---------------------------------------------------------------------------
template <int NCOLS, int NTILES>
__launch_bounds__(256)
__global__ void gemm3_wmma(const float* __restrict__ X,
                           const v16bf* __restrict__ F0,   // packed fragments
                           const v16bf* __restrict__ F1,
                           const v16bf* __restrict__ Fr,
                           const float* __restrict__ bias,
                           float* __restrict__ Y0,
                           float* __restrict__ Y1,
                           float* __restrict__ Yr,
                           int nrows) {
    const int lane = threadIdx.x & 31;
    const int wave = threadIdx.x >> 5;
    const int row0 = (blockIdx.x * (blockDim.x >> 5) + wave) * 16;
    if (row0 >= nrows) return;

    const int n    = lane & 15;
    const int half = lane >> 4;
    const int row  = row0 + n;

    v16bf a[4];
#pragma unroll
    for (int kc = 0; kc < 4; ++kc) a[kc] = load_a_frag(X, row, 128, kc * 32, lane);

    const int rowD = row0 + half * 8;              // D: VGPR r -> M = r + 8*half

#pragma unroll
    for (int w = 0; w < 3; ++w) {
        const v16bf* F = (w == 0) ? F0 : (w == 1) ? F1 : Fr;
        float*       Y = (w == 0) ? Y0 : (w == 1) ? Y1 : Yr;
#pragma unroll
        for (int nt = 0; nt < NTILES; ++nt) {
            v8f c = {};
#pragma unroll
            for (int kc = 0; kc < 4; ++kc) {
                const v16bf b = F[(size_t)(nt * 4 + kc) * 32 + lane];
                c = __builtin_amdgcn_wmma_f32_16x16x32_bf16(
                        /*neg_a=*/false, a[kc], /*neg_b=*/false, b,
                        /*c_mod=*/(short)0, c, /*reuse_a=*/false, /*reuse_b=*/false);
            }
            const int col = nt * 16 + n;
            if ((NCOLS % 16 == 0) || (col < NCOLS)) {
                float* yp = Y + (size_t)rowD * NCOLS + col;
                float bv = 0.0f;
                if (w == 2) bv = bias[col];        // compile-time-dead for w=0,1
#pragma unroll
                for (int r = 0; r < 8; ++r) {
                    float v = c[r];
                    if (w == 2) v += bv;
                    yp[(size_t)r * NCOLS] = v;     // immediate-offset stores
                }
            }
        }
    }
}

// ---------------------------------------------------------------------------
// Edge phase, 128-wide features: one wave per edge. Each lane handles 4
// consecutive floats (float4 gather = coalesced 512B per row), blends with
// linear B-spline weights, scatters with f32 atomics. Lane 0 bumps the
// per-destination edge count (reused for layer 2's mean).
// ---------------------------------------------------------------------------
__launch_bounds__(256)
__global__ void edge_scatter128(const long long* __restrict__ src,
                                const long long* __restrict__ dst,
                                const float* __restrict__ u,
                                const float* __restrict__ Y0,
                                const float* __restrict__ Y1,
                                float* __restrict__ agg,
                                float* __restrict__ cnt,
                                int nedges) {
    const int lane = threadIdx.x & 31;
    const int e    = blockIdx.x * (blockDim.x >> 5) + (threadIdx.x >> 5);
    if (e >= nedges) return;
    const int   s  = (int)src[e];
    const int   d  = (int)dst[e];
    const float uu = u[e];
    const float w0 = 1.0f - uu;

    const float4 v0 = ((const float4*)(Y0 + (size_t)s * 128))[lane];
    const float4 v1 = ((const float4*)(Y1 + (size_t)s * 128))[lane];
    float* a = agg + (size_t)d * 128 + lane * 4;
    atomicAdd(a + 0, w0 * v0.x + uu * v1.x);
    atomicAdd(a + 1, w0 * v0.y + uu * v1.y);
    atomicAdd(a + 2, w0 * v0.z + uu * v1.z);
    atomicAdd(a + 3, w0 * v0.w + uu * v1.w);
    if (lane == 0) atomicAdd(cnt + d, 1.0f);
}

// Edge phase, 40-wide (layer 2). Counts already computed.
__launch_bounds__(256)
__global__ void edge_scatter40(const long long* __restrict__ src,
                               const long long* __restrict__ dst,
                               const float* __restrict__ u,
                               const float* __restrict__ H0,
                               const float* __restrict__ H1,
                               float* __restrict__ agg,
                               int nedges) {
    const int lane = threadIdx.x & 31;
    const int e    = blockIdx.x * (blockDim.x >> 5) + (threadIdx.x >> 5);
    if (e >= nedges) return;
    const int   s  = (int)src[e];
    const int   d  = (int)dst[e];
    const float uu = u[e];
    const float w0 = 1.0f - uu;
    for (int c = lane; c < NCLS; c += 32) {
        const float m = w0 * H0[(size_t)s * NCLS + c] + uu * H1[(size_t)s * NCLS + c];
        atomicAdd(agg + (size_t)d * NCLS + c, m);
    }
}

// mean + residual(+bias already folded) + ELU -> h. float4: ncols % 4 == 0,
// so all 4 vector elements share one node / one count.
__launch_bounds__(256)
__global__ void finalize_elu4(const float4* __restrict__ agg,
                              const float* __restrict__ cnt,
                              const float4* __restrict__ xr,
                              float4* __restrict__ h, int nvec, int ncols4) {
    const int tid = blockIdx.x * blockDim.x + threadIdx.x;
    if (tid >= nvec) return;
    float c = cnt[tid / ncols4];
    c = c > 1.0f ? c : 1.0f;
    const float rc = 1.0f / c;
    const float4 a = agg[tid], x = xr[tid];
    float4 v;
    v.x = a.x * rc + x.x;  v.y = a.y * rc + x.y;
    v.z = a.z * rc + x.z;  v.w = a.w * rc + x.w;
    v.x = v.x > 0.0f ? v.x : (expf(v.x) - 1.0f);
    v.y = v.y > 0.0f ? v.y : (expf(v.y) - 1.0f);
    v.z = v.z > 0.0f ? v.z : (expf(v.z) - 1.0f);
    v.w = v.w > 0.0f ? v.w : (expf(v.w) - 1.0f);
    h[tid] = v;
}

// mean + residual -> output (float4)
__launch_bounds__(256)
__global__ void finalize_out4(const float4* __restrict__ agg,
                              const float* __restrict__ cnt,
                              const float4* __restrict__ hr,
                              float4* __restrict__ out, int nvec, int ncols4) {
    const int tid = blockIdx.x * blockDim.x + threadIdx.x;
    if (tid >= nvec) return;
    float c = cnt[tid / ncols4];
    c = c > 1.0f ? c : 1.0f;
    const float rc = 1.0f / c;
    const float4 a = agg[tid], x = hr[tid];
    float4 v;
    v.x = a.x * rc + x.x;  v.y = a.y * rc + x.y;
    v.z = a.z * rc + x.z;  v.w = a.w * rc + x.w;
    out[tid] = v;
}

// ---------------------------------------------------------------------------
extern "C" void kernel_launch(void* const* d_in, const int* in_sizes, int n_in,
                              void* d_out, int out_size, void* d_ws, size_t ws_size,
                              hipStream_t stream) {
    (void)in_sizes; (void)n_in; (void)out_size; (void)ws_size;

    const float*     x     = (const float*)d_in[0];
    const long long* eidx  = (const long long*)d_in[1];   // int64 (2, E)
    const float*     eattr = (const float*)d_in[2];       // (E, 1) -> u
    const float*     W1    = (const float*)d_in[3];       // (2,128,128)
    const float*     root1 = (const float*)d_in[4];       // (128,128)
    const float*     b1    = (const float*)d_in[5];       // (128,)
    const float*     W2    = (const float*)d_in[6];       // (2,128,40)
    const float*     root2 = (const float*)d_in[7];       // (128,40)
    const float*     b2    = (const float*)d_in[8];       // (40,)
    float*           out   = (float*)d_out;

    const long long* esrc = eidx;
    const long long* edst = eidx + NEDGES;

    // Workspace layout (floats). Peak ~257 MB; layer-2 tensors recycle the
    // y0/y1 region after layer 1's edge phase has consumed it.
    float* ws   = (float*)d_ws;
    float* y0   = ws;                               // 12.8M
    float* y1   = ws + (size_t)12800000;            // 12.8M
    float* xr   = ws + (size_t)25600000;            // 12.8M
    float* agg1 = ws + (size_t)38400000;            // 12.8M
    float* cnt  = ws + (size_t)51200000;            // 100K
    float* h    = ws + (size_t)51300000;            // 12.8M
    float* h0   = ws;                               // 4M (reuses y0)
    float* h1   = ws + (size_t)4000000;             // 4M
    float* hr   = ws + (size_t)8000000;             // 4M
    float* agg2 = ws + (size_t)12000000;            // 4M
    // Packed bf16 weight fragments (32B-aligned: float offsets are /8).
    __bf16* wp1 = (__bf16*)(ws + (size_t)64100000);        // 3 x 16384 bf16 = 96 KB
    __bf16* wp2 = (__bf16*)(ws + (size_t)64100000 + 24576);// 3 x  6144 bf16 = 36 KB
    const size_t F1SZ = (size_t)8 * 4 * 32 * 16;           // bf16 per layer-1 matrix
    const size_t F2SZ = (size_t)3 * 4 * 32 * 16;           // bf16 per layer-2 matrix

    const int rowTiles   = (NNODES + 15) / 16;                 // 6250
    const int gemmBlocks = (rowTiles + 7) / 8;                 // 8 waves/block
    const int edgeBlocks = (NEDGES + 7) / 8;                   // 8 edges/block

    // ---- One-time weight pre-swizzle (tiny) ----
    {
        const int t1 = 8 * 4 * 32, g1 = (t1 + 255) / 256;      // 1024 threads
        pack_w<<<g1, 256, 0, stream>>>(W1,                     wp1,            HID, 8);
        pack_w<<<g1, 256, 0, stream>>>(W1 + (size_t)FIN * HID, wp1 + F1SZ,     HID, 8);
        pack_w<<<g1, 256, 0, stream>>>(root1,                  wp1 + 2 * F1SZ, HID, 8);
        const int t2 = 3 * 4 * 32, g2 = (t2 + 255) / 256;      // 384 threads
        pack_w<<<g2, 256, 0, stream>>>(W2,                     wp2,            NCLS, 3);
        pack_w<<<g2, 256, 0, stream>>>(W2 + (size_t)HID * NCLS, wp2 + F2SZ,    NCLS, 3);
        pack_w<<<g2, 256, 0, stream>>>(root2,                  wp2 + 2 * F2SZ, NCLS, 3);
    }

    // ---- Layer 1 ----
    hipMemsetAsync(agg1, 0, (size_t)NNODES * HID * sizeof(float), stream);
    hipMemsetAsync(cnt,  0, (size_t)NNODES * sizeof(float), stream);

    gemm3_wmma<HID, HID / 16><<<gemmBlocks, 256, 0, stream>>>(
        x, (const v16bf*)wp1, (const v16bf*)(wp1 + F1SZ), (const v16bf*)(wp1 + 2 * F1SZ),
        b1, y0, y1, xr, NNODES);

    edge_scatter128<<<edgeBlocks, 256, 0, stream>>>(
        esrc, edst, eattr, y0, y1, agg1, cnt, NEDGES);

    {
        const int nvec = NNODES * HID / 4;
        finalize_elu4<<<(nvec + 255) / 256, 256, 0, stream>>>(
            (const float4*)agg1, cnt, (const float4*)xr, (float4*)h, nvec, HID / 4);
    }

    // ---- Layer 2 ----
    hipMemsetAsync(agg2, 0, (size_t)NNODES * NCLS * sizeof(float), stream);

    gemm3_wmma<NCLS, 3><<<gemmBlocks, 256, 0, stream>>>(
        h, (const v16bf*)wp2, (const v16bf*)(wp2 + F2SZ), (const v16bf*)(wp2 + 2 * F2SZ),
        b2, h0, h1, hr, NNODES);

    edge_scatter40<<<edgeBlocks, 256, 0, stream>>>(
        esrc, edst, eattr, h0, h1, agg2, NEDGES);

    {
        const int nvec = NNODES * NCLS / 4;
        finalize_out4<<<(nvec + 255) / 256, 256, 0, stream>>>(
            (const float4*)agg2, cnt, (const float4*)hr, (float4*)out, nvec, NCLS / 4);
    }
}
